// LinearPerformerAttention_618475290754
// MI455X (gfx1250) — matrix operand order repaired and checked by reading, not compiled, not run on hardware
//
#include <hip/hip_runtime.h>
#include <hip/hip_bf16.h>

// ---------------------------------------------------------------------------
// Types & helpers
// ---------------------------------------------------------------------------
typedef __attribute__((ext_vector_type(16))) __bf16 v16bf;
typedef __attribute__((ext_vector_type(8)))  float  v8f;
typedef __attribute__((ext_vector_type(4)))  unsigned int v4u;
typedef __attribute__((ext_vector_type(8)))  int v8i;
typedef __attribute__((ext_vector_type(4)))  int v4i;
typedef __bf16 bf16_t;

static __device__ __forceinline__ bf16_t f2bf(float f) {
    unsigned u = __float_as_uint(f);
    unsigned r = (u + 0x7FFFu + ((u >> 16) & 1u)) >> 16;   // RNE
    unsigned short s = (unsigned short)r;
    bf16_t out;
    __builtin_memcpy(&out, &s, 2);
    return out;
}
static __device__ __forceinline__ float bf2f(bf16_t b) {
    unsigned short s;
    __builtin_memcpy(&s, &b, 2);
    return __uint_as_float(((unsigned)s) << 16);
}

union FragU { v16bf v; uint4 q[2]; };

// A-fragment (16x32, MxK): lane m=lane&15, half=lane>>4.
// elements [0..7] <- K = half*8 + 0..7 ; [8..15] <- K = 16+half*8 + 0..7
static __device__ __forceinline__ v16bf load_afrag(const bf16_t* rowk0, int half) {
    FragU f;
    f.q[0] = *(const uint4*)(rowk0 + half * 8);
    f.q[1] = *(const uint4*)(rowk0 + 16 + half * 8);
    return f.v;
}
// B-fragment fed from Bt rows: elements [0..7] <- K = half*8 + 0..7 ; [8..15] <- K = 16+half*8 + 0..7
static __device__ __forceinline__ v16bf load_bfrag(const bf16_t* rowk0, int half) {
    FragU f;
    f.q[0] = *(const uint4*)(rowk0 + half * 8);
    f.q[1] = *(const uint4*)(rowk0 + 16 + half * 8);
    return f.v;
}

static __device__ __forceinline__ v8f wmma_bf16(v16bf a, v16bf b, v8f c) {
    return __builtin_amdgcn_wmma_f32_16x16x32_bf16(false, a, false, b, (short)0, c, false, false);
}

// Async global->LDS copy (16B per lane), tracked by ASYNCcnt.
static __device__ __forceinline__ void async_ld_b128(bf16_t* lds, const bf16_t* g) {
    unsigned l = (unsigned)(size_t)lds;                      // generic LDS addr[31:0] == LDS offset
    unsigned long long ga = (unsigned long long)(size_t)g;
    asm volatile("global_load_async_to_lds_b128 %0, %1, off" :: "v"(l), "v"(ga) : "memory");
}
static __device__ __forceinline__ void wait_async0() {
    asm volatile("s_wait_asynccnt 0x0" ::: "memory");
}

// ---------------------------------------------------------------------------
// Problem constants
// ---------------------------------------------------------------------------
#define PB   4
#define PN   4096
#define PD   1024
#define PH   16
#define PF   128
#define PHD  64
#define BH   (PB * PH)

// ---------------------------------------------------------------------------
// Kernel 0: proj [H][HD][F] f32 -> projT [H][F][HD] bf16
// ---------------------------------------------------------------------------
__global__ void projT_kernel(const float* __restrict__ proj, bf16_t* __restrict__ projT) {
    int h = blockIdx.x;
    for (int i = threadIdx.x; i < PHD * PF; i += 256) {
        int hd = i >> 7;
        int f  = i & (PF - 1);
        projT[(size_t)h * PF * PHD + (size_t)f * PHD + hd] =
            f2bf(proj[(size_t)h * PHD * PF + (size_t)hd * PF + f]);
    }
}

// ---------------------------------------------------------------------------
// Kernels 1/5: D[M,N] = A[M,K]*Bt[N,K]^T (+bias). 128x128 block, 8 waves 2x4,
// 64x32 per wave. Double-buffered LDS: one barrier/K-step; next tile's global
// loads are in flight under the current tile's WMMAs. fp32 sources prefetch
// through VGPRs (convert once at staging); bf16 A uses async global->LDS.
// ---------------------------------------------------------------------------
template <bool A_F32, bool OUT_F32>
__global__ __launch_bounds__(256) void gemm_bt_kernel(
    const void* __restrict__ Aptr, const float* __restrict__ Btptr,
    const float* __restrict__ bias, void* __restrict__ Dptr,
    int K, int lda, int ldb, int ldd) {
    constexpr int LSTR = 40;                    // 80B rows: 16B-aligned, conflict-padded
    __shared__ bf16_t As[2][128 * LSTR];
    __shared__ bf16_t Bs[2][128 * LSTR];

    const int lane = threadIdx.x & 31, wid = threadIdx.x >> 5;
    const int half = lane >> 4, l16 = lane & 15;
    const int wm = wid >> 2, wn = wid & 3;      // 2 x 4 wave grid
    const int rowBase = blockIdx.y * 128, colBase = blockIdx.x * 128;

    const int srow = threadIdx.x >> 1;          // staging: 2 threads/row, 16 elems each
    const int sseg = (threadIdx.x & 1) * 16;

    v8f acc[4][2] = {};
    float aReg[16];
    float bReg[16];

    auto issueA = [&](int k0, int buf) {
        if constexpr (A_F32) {
            const float4* p = (const float4*)((const float*)Aptr +
                              (size_t)(rowBase + srow) * lda + k0 + sseg);
            float4 a = p[0], b = p[1], c = p[2], d = p[3];
            aReg[0]=a.x; aReg[1]=a.y; aReg[2]=a.z;  aReg[3]=a.w;
            aReg[4]=b.x; aReg[5]=b.y; aReg[6]=b.z;  aReg[7]=b.w;
            aReg[8]=c.x; aReg[9]=c.y; aReg[10]=c.z; aReg[11]=c.w;
            aReg[12]=d.x; aReg[13]=d.y; aReg[14]=d.z; aReg[15]=d.w;
        } else {
            const bf16_t* s = (const bf16_t*)Aptr + (size_t)(rowBase + srow) * lda + k0 + sseg;
            bf16_t* d = &As[buf][srow * LSTR + sseg];
            async_ld_b128(d, s);                // ASYNCcnt-tracked, no VGPR round-trip
            async_ld_b128(d + 8, s + 8);
        }
    };
    auto issueB = [&](int k0) {
        const float4* p = (const float4*)(Btptr + (size_t)(colBase + srow) * ldb + k0 + sseg);
        float4 a = p[0], b = p[1], c = p[2], d = p[3];
        bReg[0]=a.x; bReg[1]=a.y; bReg[2]=a.z;  bReg[3]=a.w;
        bReg[4]=b.x; bReg[5]=b.y; bReg[6]=b.z;  bReg[7]=b.w;
        bReg[8]=c.x; bReg[9]=c.y; bReg[10]=c.z; bReg[11]=c.w;
        bReg[12]=d.x; bReg[13]=d.y; bReg[14]=d.z; bReg[15]=d.w;
    };
    auto commitA = [&](int buf) {
        if constexpr (A_F32) {
            bf16_t* d = &As[buf][srow * LSTR + sseg];
#pragma unroll
            for (int i = 0; i < 16; ++i) d[i] = f2bf(aReg[i]);
        }
    };
    auto commitB = [&](int buf) {
        bf16_t* d = &Bs[buf][srow * LSTR + sseg];
#pragma unroll
        for (int i = 0; i < 16; ++i) d[i] = f2bf(bReg[i]);
    };

    const int nSteps = K / 32;

    // prologue: stage tile 0 into buffer 0
    issueA(0, 0);
    issueB(0);
    commitA(0);
    commitB(0);

    for (int i = 0; i < nSteps; ++i) {
        const int cur = i & 1, nxt = cur ^ 1;
        if constexpr (!A_F32) wait_async0();    // my async copies into cur complete
        __syncthreads();                        // cur visible to all; prev reads of nxt done

        if (i + 1 < nSteps) {                   // start staging next tile under the WMMAs
            issueA((i + 1) * 32, nxt);
            issueB((i + 1) * 32);
        }

        v16bf bfr[2];
#pragma unroll
        for (int sn = 0; sn < 2; ++sn)
            bfr[sn] = load_bfrag(&Bs[cur][(wn * 32 + sn * 16 + l16) * LSTR], half);
#pragma unroll
        for (int sm = 0; sm < 4; ++sm) {
            v16bf af = load_afrag(&As[cur][(wm * 64 + sm * 16 + l16) * LSTR], half);
#pragma unroll
            for (int sn = 0; sn < 2; ++sn)
                acc[sm][sn] = wmma_bf16(af, bfr[sn], acc[sm][sn]);
        }

        if (i + 1 < nSteps) {
            commitA(nxt);
            commitB(nxt);
        }
    }

#pragma unroll
    for (int sm = 0; sm < 4; ++sm)
#pragma unroll
        for (int sn = 0; sn < 2; ++sn)
#pragma unroll
            for (int r = 0; r < 8; ++r) {
                int row = rowBase + wm * 64 + sm * 16 + r + 8 * half;
                int col = colBase + wn * 32 + sn * 16 + l16;
                float v = acc[sm][sn][r];
                if (bias) v += bias[col];
                if constexpr (OUT_F32)
                    ((float*)Dptr)[(size_t)row * ldd + col] = v;
                else
                    ((bf16_t*)Dptr)[(size_t)row * ldd + col] = f2bf(v);
            }
}

// ---------------------------------------------------------------------------
// Kernel 2: feature map. T = X[4096,64] @ projT[h]^T (K=64), elu(x)+1 ->
// q_p/k_p bf16 [BH][4096][128]. projT[h] (128x64 bf16) staged in LDS by the
// Tensor Data Mover (one tensor_load_to_lds, TENSORcnt-drained); A-fragments
// streamed straight from the qkv buffer.
// ---------------------------------------------------------------------------
__global__ __launch_bounds__(256) void feat_kernel(
    const bf16_t* __restrict__ qkv, const bf16_t* __restrict__ projT,
    bf16_t* __restrict__ q_p, bf16_t* __restrict__ k_p) {
    constexpr int PS = 72;                      // 144B rows, 16B-aligned, padded
    __shared__ bf16_t Ps[PF * PS];

    const int zz = blockIdx.y;
    const int isK = zz & 1, bh = zz >> 1;
    const int b = bh >> 4, h = bh & 15;
    const bf16_t* src = projT + (size_t)h * PF * PHD;

#if __has_builtin(__builtin_amdgcn_tensor_load_to_lds)
    if (threadIdx.x < 32) {                     // wave 0 drives the TDM
        unsigned ldsA = (unsigned)(size_t)&Ps[0];
        unsigned long long ga = (unsigned long long)(size_t)src;
        // D# group0: count=1, lds_addr, global_addr, type=2
        v4u g0 = { 1u, ldsA, (unsigned)ga, (unsigned)(ga >> 32) | (2u << 30) };
        // D# group1: data_size=2B(code1), pad_enable, pad_interval=32dw(code4),
        // pad_amount=4dw(code3); tensor 64x128, tile 64x128, dim0_stride=64
        unsigned w0 = (1u << 16) | (1u << 20) | (4u << 22) | (3u << 25);
        unsigned w1 = (unsigned)(PHD & 0xFFFF) << 16;          // tensor_dim0 lo
        unsigned w2 = (unsigned)(PF & 0xFFFF) << 16;           // tensor_dim1 lo
        unsigned w3 = (unsigned)(PHD & 0xFFFF) << 16;          // tile_dim0
        unsigned w4 = (unsigned)(PF & 0xFFFF);                 // tile_dim1
        unsigned w5 = (unsigned)PHD;                           // dim0_stride lo
        v8i g1 = { (int)w0, (int)w1, (int)w2, (int)w3, (int)w4, (int)w5, 0, 0 };
        v4i gz4 = { 0, 0, 0, 0 };
        v8i gz8 = { 0, 0, 0, 0, 0, 0, 0, 0 };
        __builtin_amdgcn_tensor_load_to_lds(g0, g1, gz4, gz4, gz8, 0);
        __builtin_amdgcn_s_wait_tensorcnt(0);
    }
#else
    {
        int row = threadIdx.x >> 1, seg = (threadIdx.x & 1) * 32;
#pragma unroll
        for (int i = 0; i < 4; ++i)
            ((uint4*)(Ps + row * PS + seg))[i] = ((const uint4*)(src + row * PHD + seg))[i];
    }
#endif
    __syncthreads();

    const int lane = threadIdx.x & 31, wid = threadIdx.x >> 5;
    const int half = lane >> 4, l16 = lane & 15;
    const int nFrag = blockIdx.x * 128 + wid * 16 + l16;

    const bf16_t* arow = qkv + ((size_t)(b * PN + nFrag)) * (3 * PD) + h * PHD + (isK ? PD : 0);

    v8f acc[8] = {};
#pragma unroll
    for (int k0 = 0; k0 < PHD; k0 += 32) {
        v16bf af = load_afrag(arow + k0, half);
#pragma unroll
        for (int sn = 0; sn < 8; ++sn) {
            v16bf bfr = load_bfrag(Ps + (sn * 16 + l16) * PS + k0, half);
            acc[sn] = wmma_bf16(af, bfr, acc[sn]);
        }
    }

    bf16_t* dst = (isK ? k_p : q_p) + (size_t)bh * PN * PF;
#pragma unroll
    for (int sn = 0; sn < 8; ++sn)
#pragma unroll
        for (int r = 0; r < 8; ++r) {
            int row = blockIdx.x * 128 + wid * 16 + r + 8 * half;
            int col = sn * 16 + l16;
            float v = acc[sn][r];
            v = (v > 0.f) ? (v + 1.f) : __expf(v);   // elu(x)+1
            dst[(size_t)row * PF + col] = f2bf(v);
        }
}

// ---------------------------------------------------------------------------
// Kernel 3: kv[f][hd] accumulation (split-K over sequence) + k_sum.
// grid (BH, 4). Block tile 128(f) x 64(hd); 8 waves 4x2, 32x32 each.
// LDS-transposed staging so both fragments read contiguous rows.
// ---------------------------------------------------------------------------
__global__ __launch_bounds__(256) void kv_kernel(
    const bf16_t* __restrict__ qkv, const bf16_t* __restrict__ k_p,
    float* __restrict__ kvT, float* __restrict__ k_sum) {
    constexpr int TS = 40;
    __shared__ bf16_t Apt[PF * TS];             // [f][n]
    __shared__ bf16_t Vt[PHD * TS];             // [hd][n]

    const int bh = blockIdx.x;
    const int b = bh >> 4, h = bh & 15;
    const int nBase = blockIdx.y * 1024;

    const int lane = threadIdx.x & 31, wid = threadIdx.x >> 5;
    const int half = lane >> 4, l16 = lane & 15;
    const int wm = wid >> 1, wn = wid & 1;

    v8f acc[2][2] = {};

    for (int nn = 0; nn < 1024; nn += 32) {
        __syncthreads();
        const int n0 = nBase + nn;
        {   // k_p tile [32n x 128f] -> Apt[f][n]
            int nrow = threadIdx.x >> 3, f0 = (threadIdx.x & 7) * 16;
            const bf16_t* s = k_p + ((size_t)bh * PN + n0 + nrow) * PF + f0;
            FragU t;
            t.q[0] = ((const uint4*)s)[0];
            t.q[1] = ((const uint4*)s)[1];
#pragma unroll
            for (int i = 0; i < 16; ++i) Apt[(f0 + i) * TS + nrow] = t.v[i];
        }
        if (threadIdx.x < 128) {                // v tile [32n x 64hd] -> Vt[hd][n]
            int nrow = threadIdx.x >> 2, hd0 = (threadIdx.x & 3) * 16;
            const bf16_t* s = qkv + ((size_t)(b * PN + n0 + nrow)) * (3 * PD) + 2 * PD + h * PHD + hd0;
            FragU t;
            t.q[0] = ((const uint4*)s)[0];
            t.q[1] = ((const uint4*)s)[1];
#pragma unroll
            for (int i = 0; i < 16; ++i) Vt[(hd0 + i) * TS + nrow] = t.v[i];
        }
        __syncthreads();

        v16bf af[2], bfr[2];
#pragma unroll
        for (int sm = 0; sm < 2; ++sm)
            af[sm] = load_afrag(Apt + (wm * 32 + sm * 16 + l16) * TS, half);
#pragma unroll
        for (int sn = 0; sn < 2; ++sn)
            bfr[sn] = load_bfrag(Vt + (wn * 32 + sn * 16 + l16) * TS, half);
#pragma unroll
        for (int sm = 0; sm < 2; ++sm)
#pragma unroll
            for (int sn = 0; sn < 2; ++sn)
                acc[sm][sn] = wmma_bf16(af[sm], bfr[sn], acc[sm][sn]);
    }

#pragma unroll
    for (int sm = 0; sm < 2; ++sm)
#pragma unroll
        for (int sn = 0; sn < 2; ++sn)
#pragma unroll
            for (int r = 0; r < 8; ++r) {
                int f  = wm * 32 + sm * 16 + r + 8 * half;
                int hd = wn * 32 + sn * 16 + l16;
                atomicAdd(&kvT[(size_t)bh * PHD * PF + (size_t)hd * PF + f], acc[sm][sn][r]);
            }

    if (threadIdx.x < PF) {                     // k_sum partial for this chunk
        int f = threadIdx.x;
        const bf16_t* kp = k_p + ((size_t)bh * PN + nBase) * PF + f;
        float s = 0.f;
        for (int i = 0; i < 1024; ++i) s += bf2f(kp[(size_t)i * PF]);
        atomicAdd(&k_sum[bh * PF + f], s);
    }
}

// ---------------------------------------------------------------------------
// Kernel 4: attn = normalize(q_p @ kv), head-interleaved bf16.
// grid (N/128, BH); block tile 128(n) x 64(hd); kvT staged f32->bf16 in LDS.
// ---------------------------------------------------------------------------
__global__ __launch_bounds__(256) void attn_kernel(
    const bf16_t* __restrict__ q_p, const float* __restrict__ kvT,
    const float* __restrict__ k_sum, bf16_t* __restrict__ attn) {
    constexpr int BS = 136;                     // 272B rows, 16B-aligned, padded
    __shared__ bf16_t Bs[PHD * BS];
    __shared__ float zsh[128];
    __shared__ float ksh[PF];

    const int bh = blockIdx.y;
    const int b = bh >> 4, h = bh & 15;
    const int n0 = blockIdx.x * 128;

    {   // stage kvT[bh] (64 x 128 f32) -> bf16 LDS
        int hd = threadIdx.x >> 2, f0 = (threadIdx.x & 3) * 32;
        const float* s = kvT + (size_t)bh * PHD * PF + (size_t)hd * PF + f0;
#pragma unroll
        for (int i = 0; i < 32; ++i) Bs[hd * BS + f0 + i] = f2bf(s[i]);
    }
    if (threadIdx.x < PF) ksh[threadIdx.x] = k_sum[bh * PF + threadIdx.x];
    __syncthreads();

    if (threadIdx.x < 128) {                    // per-row normalizer z
        int n = n0 + threadIdx.x;
        const bf16_t* q = q_p + ((size_t)bh * PN + n) * PF;
        float s = 1e-8f;
        for (int f = 0; f < PF; ++f) s += bf2f(q[f]) * ksh[f];
        zsh[threadIdx.x] = 1.f / s;
    }
    __syncthreads();

    const int lane = threadIdx.x & 31, wid = threadIdx.x >> 5;
    const int half = lane >> 4, l16 = lane & 15;
    const int wm = wid >> 1, wn = wid & 1;

    v8f acc[2][2] = {};
#pragma unroll
    for (int k0 = 0; k0 < PF; k0 += 32) {
        v16bf af[2], bfr[2];
#pragma unroll
        for (int sm = 0; sm < 2; ++sm) {
            int n = n0 + wm * 32 + sm * 16 + l16;
            af[sm] = load_afrag(q_p + ((size_t)bh * PN + n) * PF + k0, half);
        }
#pragma unroll
        for (int sn = 0; sn < 2; ++sn)
            bfr[sn] = load_bfrag(Bs + (wn * 32 + sn * 16 + l16) * BS + k0, half);
#pragma unroll
        for (int sm = 0; sm < 2; ++sm)
#pragma unroll
            for (int sn = 0; sn < 2; ++sn)
                acc[sm][sn] = wmma_bf16(af[sm], bfr[sn], acc[sm][sn]);
    }

#pragma unroll
    for (int sm = 0; sm < 2; ++sm)
#pragma unroll
        for (int sn = 0; sn < 2; ++sn)
#pragma unroll
            for (int r = 0; r < 8; ++r) {
                int lr = wm * 32 + sm * 16 + r + 8 * half;
                int n = n0 + lr;
                int hd = wn * 32 + sn * 16 + l16;
                float v = acc[sm][sn][r] * zsh[lr];
                attn[((size_t)(b * PN + n)) * PD + h * PHD + hd] = f2bf(v);
            }
}

// ---------------------------------------------------------------------------
// Host launcher
// ---------------------------------------------------------------------------
extern "C" void kernel_launch(void* const* d_in, const int* in_sizes, int n_in,
                              void* d_out, int out_size, void* d_ws, size_t ws_size,
                              hipStream_t stream) {
    const float* x     = (const float*)d_in[0];  // [B,N,D]
    const float* W_qkv = (const float*)d_in[1];  // [3D,D]
    const float* proj  = (const float*)d_in[2];  // [H,HD,F]
    const float* W_out = (const float*)d_in[3];  // [D,D]
    const float* b_out = (const float*)d_in[4];  // [D]
    float* out = (float*)d_out;

    const size_t M = (size_t)PB * PN;                      // 16384
    const size_t qkv_off   = 0;
    const size_t qkv_sz    = M * 3 * PD * 2;               // bf16 ~100.7MB
    const size_t qp_off    = qkv_off + qkv_sz;
    const size_t qp_sz     = (size_t)BH * PN * PF * 2;     // bf16 67MB
    const size_t kp_off    = qp_off + qp_sz;
    const size_t kp_sz     = qp_sz;                        // bf16 67MB
    const size_t attn_off  = kp_off;                       // alias: k_p dead after kv_kernel
    const size_t projT_off = kp_off + kp_sz;
    const size_t projT_sz  = (size_t)PH * PF * PHD * 2;    // 256KB
    const size_t kvT_off   = projT_off + projT_sz;
    const size_t kvT_sz    = (size_t)BH * PHD * PF * 4;    // f32 2MB
    const size_t ksum_off  = kvT_off + kvT_sz;
    const size_t ksum_sz   = (size_t)BH * PF * 4;          // 32KB

    bf16_t* qkv   = (bf16_t*)((char*)d_ws + qkv_off);
    bf16_t* q_p   = (bf16_t*)((char*)d_ws + qp_off);
    bf16_t* k_p   = (bf16_t*)((char*)d_ws + kp_off);
    bf16_t* attn  = (bf16_t*)((char*)d_ws + attn_off);
    bf16_t* projT = (bf16_t*)((char*)d_ws + projT_off);
    float*  kvT   = (float*)((char*)d_ws + kvT_off);
    float*  ksum  = (float*)((char*)d_ws + ksum_off);

    (void)hipMemsetAsync((char*)d_ws + kvT_off, 0, kvT_sz + ksum_sz, stream);

    projT_kernel<<<PH, 256, 0, stream>>>(proj, projT);

    // 1) qkv = x @ W_qkv^T  (M=16384, N=3072, K=1024), bf16 out
    gemm_bt_kernel<true, false><<<dim3(3 * PD / 128, M / 128), 256, 0, stream>>>(
        x, W_qkv, nullptr, qkv, PD, PD, PD, 3 * PD);

    // 2) feature maps q_p / k_p
    feat_kernel<<<dim3(PN / 128, BH * 2), 256, 0, stream>>>(qkv, projT, q_p, k_p);

    // 3) kv accumulation + k_sum (split-K over sequence, 4 chunks)
    kv_kernel<<<dim3(BH, 4), 256, 0, stream>>>(qkv, k_p, kvT, ksum);

    // 4) attn = normalize(q_p @ kv)
    attn_kernel<<<dim3(PN / 128, BH), 256, 0, stream>>>(q_p, kvT, ksum, attn);

    // 5) out = attn @ W_out^T + b_out  (f32 out, async-LDS A staging)
    gemm_bt_kernel<false, true><<<dim3(PD / 128, M / 128), 256, 0, stream>>>(
        attn, W_out, b_out, out, PD, PD, PD, PD);
}
